// EntropyByteLatentTransformer_18657337933964
// MI455X (gfx1250) — matrix operand order repaired and checked by reading, not compiled, not run on hardware
//
#include <hip/hip_runtime.h>
#include <hip/hip_fp16.h>

// ---------------- model constants ----------------
#define DMODEL 512
#define NHEAD  8
#define NLAYER 8
#define VOCAB  258
#define SEQ    1024
#define BATCH  8
#define MTOK   (BATCH * SEQ)   // 8192 tokens
#define DFFN   2048
#define DHEAD  64
#define WIN    8

typedef _Float16 f16;
typedef f16   v16h __attribute__((ext_vector_type(16)));
typedef f16   v8h  __attribute__((ext_vector_type(8)));
typedef float v8f  __attribute__((ext_vector_type(8)));
typedef unsigned int u32;
typedef u32 u32x4 __attribute__((ext_vector_type(4)));
typedef int  i32x8 __attribute__((ext_vector_type(8)));
typedef int  i32x4 __attribute__((ext_vector_type(4)));

__device__ __forceinline__ v16h cat16(v8h lo, v8h hi) {
  return __builtin_shufflevector(lo, hi, 0,1,2,3,4,5,6,7,8,9,10,11,12,13,14,15);
}
__device__ __forceinline__ v8f wmma_f32_f16(v16h a, v16h b, v8f c) {
  // D = A(16x32 f16) * B(32x16 f16) + C(16x16 f32)
  return __builtin_amdgcn_wmma_f32_16x16x32_f16(false, a, false, b, (short)0, c,
                                                false, false);
}
__device__ __forceinline__ v8f zero8() {
  v8f z = {0.f,0.f,0.f,0.f,0.f,0.f,0.f,0.f};
  return z;
}

// ---- Tensor Data Mover: load a 64-row x 32-col f16 tile (rows stride K elems)
// into LDS at lds_off, with hardware padding: 16B pad after every 64B stored
// -> 80B LDS row pitch (bank-conflict-free b128 reads, still 16B aligned).
// D# per CDNA5 ISA ch.8; 2-D tensor so groups 2/3 (and the unused 5th SGPR
// group of the 6-arg builtin) are zero-filled.
#define BSTAGE_PITCH 80
#define BSTAGE_BYTES (64 * BSTAGE_PITCH)   // 5120 per buffer
__device__ __forceinline__ void tdm_load_b_tile(u32 lds_off, const f16* gptr,
                                                u32 k_elems, u32 n_rows_remaining) {
  unsigned long long ga = (unsigned long long)(uintptr_t)gptr;
  u32x4 g0;
  g0.x = 1u;                                           // count=1 (valid user D#)
  g0.y = lds_off;                                      // lds_addr (bytes)
  g0.z = (u32)(ga & 0xffffffffu);                      // global_addr[31:0]
  g0.w = (u32)((ga >> 32) & 0x01ffffffu) | (2u << 30); // addr[56:32] | type=2
  i32x8 g1;
  g1[0] = (int)(0x00010000u      // data_size=1 -> 2 bytes
              | (1u << 20)       // pad_enable
              | (3u << 22)       // pad_interval: 16 DWORDs (64B) between pads
              | (3u << 25));     // pad_amount: 4 DWORDs (16B)
  g1[1] = (int)((k_elems & 0xffffu) << 16);                       // tensor_dim0 lo
  g1[2] = (int)(((k_elems >> 16) & 0xffffu) |
                ((n_rows_remaining & 0xffffu) << 16));            // td0 hi | td1 lo
  g1[3] = (int)(((n_rows_remaining >> 16) & 0xffffu) |
                (32u << 16));                                     // td1 hi | tile_dim0=32
  g1[4] = (int)64u;                                               // tile_dim1=64, tile_dim2=0
  g1[5] = (int)k_elems;                                           // tensor_dim0_stride lo32
  g1[6] = 0;                                                      // stride hi | dim1_stride lo
  g1[7] = 0;
  i32x4 z4 = {0, 0, 0, 0};
  i32x8 z8 = {0, 0, 0, 0, 0, 0, 0, 0};
  __builtin_amdgcn_tensor_load_to_lds(g0, g1, z4, z4, z8, 0);
}

// ---------------- entropy features ----------------
__global__ __launch_bounds__(256) void entropy_kernel(const int* __restrict__ x,
                                                      float* __restrict__ ent) {
  int idx = blockIdx.x * blockDim.x + threadIdx.x;
  if (idx >= MTOK) return;
  int l = idx & (SEQ - 1);
  if (l > SEQ - WIN) { ent[idx] = 0.f; return; }
  int w[WIN];
  #pragma unroll
  for (int i = 0; i < WIN; ++i) w[i] = x[idx + i];
  float e = 0.f;
  int nd = 0;
  #pragma unroll
  for (int i = 0; i < WIN; ++i) {
    if (w[i] >= 256) continue;
    int c = 0; bool first = true;
    #pragma unroll
    for (int j = 0; j < WIN; ++j) {
      if (w[j] == w[i]) { c++; if (j < i) first = false; }
    }
    if (first) nd++;
    e -= 0.125f * log2f((float)c * 0.125f);
  }
  e += (float)(VOCAB - nd) * 3.3219281e-9f;   // clip(1e-10) zero-bin term
  ent[idx] = e;
}

// h[tok,d] = embed[x[tok],d] + ent[tok]*ep_w[d] + ep_b[d]
__global__ __launch_bounds__(256) void embed_kernel(const int* __restrict__ x,
                                                    const float* __restrict__ emb,
                                                    const float* __restrict__ ep_w,
                                                    const float* __restrict__ ep_b,
                                                    const float* __restrict__ ent,
                                                    float* __restrict__ h) {
  int idx = blockIdx.x * blockDim.x + threadIdx.x;
  int tok = idx >> 9, d = idx & (DMODEL - 1);
  h[idx] = emb[(size_t)x[tok] * DMODEL + d] + ent[tok] * ep_w[d] + ep_b[d];
}

// ---------------- fp32 -> f16 convert (grid-stride) ----------------
__global__ __launch_bounds__(256) void cvt_kernel(const float* __restrict__ in,
                                                  __half* __restrict__ outh, int n) {
  f16* out = (f16*)outh;
  int stride = gridDim.x * blockDim.x;
  for (int i = blockIdx.x * blockDim.x + threadIdx.x; i < n; i += stride)
    out[i] = (f16)in[i];
}

// ---------------- layernorm (one wave per token) ----------------
__global__ __launch_bounds__(256) void ln_kernel(const float* __restrict__ x,
                                                 const float* __restrict__ w,
                                                 const float* __restrict__ b,
                                                 __half* __restrict__ outh) {
  f16* out = (f16*)outh;
  int lane = threadIdx.x & 31, wave = threadIdx.x >> 5;
  int token = blockIdx.x * 8 + wave;
  const float* row = x + (size_t)token * DMODEL;
  float4 vals[4];
  float s = 0.f, s2 = 0.f;
  #pragma unroll
  for (int i = 0; i < 4; ++i) {
    vals[i] = *(const float4*)(row + i * 128 + lane * 4);
    s  += vals[i].x + vals[i].y + vals[i].z + vals[i].w;
    s2 += vals[i].x * vals[i].x + vals[i].y * vals[i].y +
          vals[i].z * vals[i].z + vals[i].w * vals[i].w;
  }
  #pragma unroll
  for (int d = 1; d < 32; d <<= 1) {
    s  += __shfl_xor(s, d, 32);
    s2 += __shfl_xor(s2, d, 32);
  }
  float mu   = s  * (1.f / DMODEL);
  float var  = s2 * (1.f / DMODEL) - mu * mu;
  float rstd = rsqrtf(var + 1e-5f);
  #pragma unroll
  for (int i = 0; i < 4; ++i) {
    float vv[4] = {vals[i].x, vals[i].y, vals[i].z, vals[i].w};
    #pragma unroll
    for (int j = 0; j < 4; ++j) {
      int d = i * 128 + lane * 4 + j;
      out[(size_t)token * DMODEL + d] = (f16)((vv[j] - mu) * rstd * w[d] + b[d]);
    }
  }
}

// ---------------- qkv repack: [M,1536] -> q/k [b,h,l,dh], vT [b,h,dh,l] ------
__global__ __launch_bounds__(256) void repack_kernel(const __half* __restrict__ qkvh,
                                                     __half* __restrict__ q,
                                                     __half* __restrict__ k,
                                                     __half* __restrict__ vT) {
  int idx = blockIdx.x * blockDim.x + threadIdx.x;   // over MTOK*1536
  int tok = idx / (3 * DMODEL), c = idx % (3 * DMODEL);
  int which = c >> 9, d = c & (DMODEL - 1);
  int head = d >> 6, dh = d & (DHEAD - 1);
  int b = tok >> 10, l = tok & (SEQ - 1);
  __half val = qkvh[idx];
  int bh = b * NHEAD + head;
  if (which == 0)      q [((size_t)bh * SEQ + l) * DHEAD + dh] = val;
  else if (which == 1) k [((size_t)bh * SEQ + l) * DHEAD + dh] = val;
  else                 vT[((size_t)bh * DHEAD + dh) * SEQ + l] = val;
}

// ---------------- WMMA GEMM with TDM-staged weights -------------------------
// Y = A(f16 MxK) @ W(f16 NxK)^T + bias. Block = 4 waves, 64(m) x 64(n) tile.
// Wave0 streams the block's 64xK weight panel through LDS in 64x32 tiles via
// tensor_load_to_lds (double-buffered, TENSORcnt); all waves read B-fragments
// from LDS (80B row pitch from TDM pad => conflict-free ds_load_b128).
enum { EPI_F16 = 0, EPI_RESID = 1, EPI_GELU = 2, EPI_F32 = 3 };

template <int EPI>
__global__ __launch_bounds__(128)
void gemm_kernel(const __half* __restrict__ Ah, const __half* __restrict__ Wh,
                 const float* __restrict__ bias, __half* __restrict__ outH,
                 float* __restrict__ outF, int M, int N, int K) {
  __shared__ __align__(16) char bstage[2][BSTAGE_BYTES];  // only LDS object -> offset 0
  const f16* A = (const f16*)Ah;
  const f16* W = (const f16*)Wh;
  int lane = threadIdx.x & 31, wave = threadIdx.x >> 5;
  int g = lane >> 4, r = lane & 15;                 // half-wave, sub-lane
  int m0 = blockIdx.x * 64 + wave * 16;
  int n0 = blockIdx.y * 64;
  bool leader = (wave == 0);
  u32 nrem = (u32)(N - n0);                         // hardware OOB-zero for tail

  v8f c[4];
  #pragma unroll
  for (int t = 0; t < 4; ++t) c[t] = zero8();

  const f16* arow  = A + (size_t)(m0 + r) * K;
  const f16* wpanel = W + (size_t)n0 * K;
  const int KS = K / 32;

  if (leader) tdm_load_b_tile(0u, wpanel, (u32)K, nrem);

  for (int ks = 0; ks < KS; ++ks) {
    if (leader) {
      if (ks + 1 < KS) {
        tdm_load_b_tile((u32)(((ks + 1) & 1) * BSTAGE_BYTES),
                        wpanel + (ks + 1) * 32, (u32)K, nrem);
        __builtin_amdgcn_s_wait_tensorcnt(1);       // current tile landed
      } else {
        __builtin_amdgcn_s_wait_tensorcnt(0);
      }
    }
    __syncthreads();                                // tile visible to all waves

    int kk = ks * 32;
    // A 16x32 fragment: lane m = r; k = (h>>3)*16 + g*8 + (h&7)
    v16h a = cat16(*(const v8h*)(arow + kk + g * 8),
                   *(const v8h*)(arow + kk + 16 + g * 8));
    const char* bs = bstage[ks & 1];
    #pragma unroll
    for (int t = 0; t < 4; ++t) {
      // B 32x16 fragment from LDS: lane n = t*16+r; k = (h>>3)*16 + g*8 + (h&7)
      const f16* wp = (const f16*)(bs + (t * 16 + r) * BSTAGE_PITCH + g * 16);
      v16h bfrag = cat16(*(const v8h*)wp, *(const v8h*)(wp + 16));
      c[t] = wmma_f32_f16(a, bfrag, c[t]);
    }
    __syncthreads();                                // reads done before overwrite
  }

  #pragma unroll
  for (int t = 0; t < 4; ++t) {
    int n = n0 + t * 16 + r;
    bool nok = (n < N);
    float bv = (bias != nullptr && nok) ? bias[n] : 0.f;
    #pragma unroll
    for (int v = 0; v < 8; ++v) {
      int m = m0 + g * 8 + v;                       // C layout: m = half*8 + vgpr
      float y = c[t][v] + bv;
      if (EPI == EPI_F16) {
        ((f16*)outH)[(size_t)m * N + n] = (f16)y;
      } else if (EPI == EPI_RESID) {
        float* p = outF + (size_t)m * N + n;
        *p = *p + y;
      } else if (EPI == EPI_GELU) {
        float gl = 0.5f * y * (1.f + erff(y * 0.70710678f));
        ((f16*)outH)[(size_t)m * N + n] = (f16)gl;
      } else {                                      // EPI_F32: logits, N=258
        if (nok) outF[(size_t)m * N + n] = y;
      }
    }
  }
}

// ---------------- flash attention: one wave = (b,h, 16-query tile) ----------
__global__ __launch_bounds__(128)
void attn_kernel(const __half* __restrict__ qh, const __half* __restrict__ kh,
                 const __half* __restrict__ vTh, __half* __restrict__ oh) {
  __shared__ __align__(16) f16 pbuf_all[4][16 * 32];
  const f16* q  = (const f16*)qh;
  const f16* kk = (const f16*)kh;
  const f16* vT = (const f16*)vTh;
  f16* o = (f16*)oh;

  int lane = threadIdx.x & 31, wave = threadIdx.x >> 5;
  int g = lane >> 4, r = lane & 15;
  int wid = blockIdx.x * 4 + wave;                  // 0 .. B*H*(L/16)-1
  int qt = wid & 63;                                // query tile
  int bh = wid >> 6;                                // b*H + h
  int b = bh >> 3, hh = bh & 7;

  const f16* qbase = q  + ((size_t)bh * SEQ + qt * 16) * DHEAD;
  const f16* kbase = kk + (size_t)bh * SEQ * DHEAD;
  const f16* vbase = vT + (size_t)bh * DHEAD * SEQ;
  f16* pb = pbuf_all[wave];

  // Q fragments covering dh 0..31 / 32..63
  v16h qa[2];
  #pragma unroll
  for (int f = 0; f < 2; ++f) {
    const f16* p = qbase + r * DHEAD + f * 32 + g * 8;
    qa[f] = cat16(*(const v8h*)p, *(const v8h*)(p + 16));
  }

  float Mr[8], Lr[8];
  #pragma unroll
  for (int v = 0; v < 8; ++v) { Mr[v] = -1e30f; Lr[v] = 0.f; }
  v8f oacc[4];
  #pragma unroll
  for (int t = 0; t < 4; ++t) oacc[t] = zero8();

  for (int jc = 0; jc < SEQ / 32; ++jc) {           // 32-key chunks
    v8f s[2];
    #pragma unroll
    for (int sub = 0; sub < 2; ++sub) {
      int j0 = jc * 32 + sub * 16;
      const f16* kp = kbase + (size_t)(j0 + r) * DHEAD;  // lane n = key
      v16h kb0 = cat16(*(const v8h*)(kp + g * 8),       // dh 0..31
                       *(const v8h*)(kp + g * 8 + 16));
      v16h kb1 = cat16(*(const v8h*)(kp + 32 + g * 8),  // dh 32..63
                       *(const v8h*)(kp + 32 + g * 8 + 16));
      v8f z = zero8();
      z = wmma_f32_f16(qa[0], kb0, z);
      z = wmma_f32_f16(qa[1], kb1, z);
      s[sub] = z;
    }
    // online softmax over this 32-key chunk; C layout: row m = g*8+v, col = r
    float scale[8];
    #pragma unroll
    for (int v = 0; v < 8; ++v) {
      float a0 = s[0][v] * 0.125f;                  // 1/sqrt(64)
      float a1 = s[1][v] * 0.125f;
      float mx = fmaxf(a0, a1);
      #pragma unroll
      for (int d = 1; d < 16; d <<= 1) mx = fmaxf(mx, __shfl_xor(mx, d, 32));
      float newM = fmaxf(Mr[v], mx);
      float sc = __expf(Mr[v] - newM);
      Mr[v] = newM;
      float p0 = __expf(a0 - newM), p1 = __expf(a1 - newM);
      float rs = p0 + p1;
      #pragma unroll
      for (int d = 1; d < 16; d <<= 1) rs += __shfl_xor(rs, d, 32);
      Lr[v] = Lr[v] * sc + rs;
      scale[v] = sc;
      int m = g * 8 + v;
      pb[m * 32 + r]      = (f16)p0;                // P tile -> LDS (C layout)
      pb[m * 32 + 16 + r] = (f16)p1;
    }
    #pragma unroll
    for (int t = 0; t < 4; ++t)
      #pragma unroll
      for (int v = 0; v < 8; ++v) oacc[t][v] *= scale[v];

    asm volatile("s_wait_dscnt 0" ::: "memory");    // P stores visible to reads

    // re-read P as A 16x32 fragment: m = r; k = (h>>3)*16 + g*8 + (h&7)
    const f16* pr = pb + r * 32 + g * 8;
    v16h pa = cat16(*(const v8h*)pr, *(const v8h*)(pr + 16));

    #pragma unroll
    for (int t = 0; t < 4; ++t) {                   // dh tiles of 16
      const f16* vp = vbase + (size_t)(t * 16 + r) * SEQ + jc * 32 + g * 8;
      v16h vb = cat16(*(const v8h*)vp, *(const v8h*)(vp + 16));
      oacc[t] = wmma_f32_f16(pa, vb, oacc[t]);
    }
  }
  // normalize and write o as f16 [token, D] (col = h*64 + dh)
  #pragma unroll
  for (int t = 0; t < 4; ++t)
    #pragma unroll
    for (int v = 0; v < 8; ++v) {
      int m = g * 8 + v;
      int tok = b * SEQ + qt * 16 + m;
      int d = hh * DHEAD + t * 16 + r;
      o[(size_t)tok * DMODEL + d] = (f16)(oacc[t][v] / Lr[v]);
    }
}

// ---------------- host orchestration ----------------
extern "C" void kernel_launch(void* const* d_in, const int* in_sizes, int n_in,
                              void* d_out, int out_size, void* d_ws, size_t ws_size,
                              hipStream_t stream) {
  (void)in_sizes; (void)n_in; (void)out_size; (void)ws_size;
  const int*   x     = (const int*)  d_in[0];
  const float* emb   = (const float*)d_in[1];
  const float* ep_w  = (const float*)d_in[2];
  const float* ep_b  = (const float*)d_in[3];
  const float* in_w  = (const float*)d_in[4];
  const float* in_b  = (const float*)d_in[5];
  const float* op_w  = (const float*)d_in[6];
  const float* op_b  = (const float*)d_in[7];
  const float* ln1_w = (const float*)d_in[8];
  const float* ln1_b = (const float*)d_in[9];
  const float* ln2_w = (const float*)d_in[10];
  const float* ln2_b = (const float*)d_in[11];
  const float* f1_w  = (const float*)d_in[12];
  const float* f1_b  = (const float*)d_in[13];
  const float* f2_w  = (const float*)d_in[14];
  const float* f2_b  = (const float*)d_in[15];
  const float* out_w = (const float*)d_in[16];

  char* ws = (char*)d_ws;
  size_t off = 0;
  auto alloc = [&](size_t bytes) -> void* {
    off = (off + 255) & ~(size_t)255;
    void* p = ws + off;
    off += bytes;
    return p;
  };
  // ~86 MB workspace
  float*  h    = (float*) alloc((size_t)MTOK * DMODEL * 4);  // residual stream
  __half* g16  = (__half*)alloc((size_t)MTOK * DMODEL * 2);  // LN out / attn out / h16
  __half* big  = (__half*)alloc((size_t)MTOK * DFFN   * 2);  // qkv (Mx1536) / ffn mid
  __half* q16  = (__half*)alloc((size_t)MTOK * DMODEL * 2);
  __half* k16  = (__half*)alloc((size_t)MTOK * DMODEL * 2);
  __half* vT16 = (__half*)alloc((size_t)MTOK * DMODEL * 2);
  float*  ent  = (float*) alloc((size_t)MTOK * 4);
  __half* wbuf = (__half*)alloc((size_t)DFFN * DMODEL * 2);  // per-use f16 weights

  entropy_kernel<<<MTOK / 256, 256, 0, stream>>>(x, ent);
  embed_kernel<<<(MTOK * DMODEL) / 256, 256, 0, stream>>>(x, emb, ep_w, ep_b, ent, h);

  for (int l = 0; l < NLAYER; ++l) {
    // --- attention ---
    ln_kernel<<<MTOK / 8, 256, 0, stream>>>(h, ln1_w + l * DMODEL, ln1_b + l * DMODEL, g16);
    cvt_kernel<<<1024, 256, 0, stream>>>(in_w + (size_t)l * 3 * DMODEL * DMODEL, wbuf,
                                         3 * DMODEL * DMODEL);
    gemm_kernel<EPI_F16><<<dim3(MTOK / 64, (3 * DMODEL) / 64), 128, 0, stream>>>(
        g16, wbuf, in_b + l * 3 * DMODEL, big, nullptr, MTOK, 3 * DMODEL, DMODEL);
    repack_kernel<<<(MTOK * 3 * DMODEL) / 256, 256, 0, stream>>>(big, q16, k16, vT16);
    attn_kernel<<<(BATCH * NHEAD * (SEQ / 16)) / 4, 128, 0, stream>>>(q16, k16, vT16, g16);
    cvt_kernel<<<1024, 256, 0, stream>>>(op_w + (size_t)l * DMODEL * DMODEL, wbuf,
                                         DMODEL * DMODEL);
    gemm_kernel<EPI_RESID><<<dim3(MTOK / 64, DMODEL / 64), 128, 0, stream>>>(
        g16, wbuf, op_b + l * DMODEL, nullptr, h, MTOK, DMODEL, DMODEL);
    // --- FFN ---
    ln_kernel<<<MTOK / 8, 256, 0, stream>>>(h, ln2_w + l * DMODEL, ln2_b + l * DMODEL, g16);
    cvt_kernel<<<1024, 256, 0, stream>>>(f1_w + (size_t)l * DFFN * DMODEL, wbuf,
                                         DFFN * DMODEL);
    gemm_kernel<EPI_GELU><<<dim3(MTOK / 64, DFFN / 64), 128, 0, stream>>>(
        g16, wbuf, f1_b + l * DFFN, big, nullptr, MTOK, DFFN, DMODEL);
    cvt_kernel<<<1024, 256, 0, stream>>>(f2_w + (size_t)l * DMODEL * DFFN, wbuf,
                                         DMODEL * DFFN);
    gemm_kernel<EPI_RESID><<<dim3(MTOK / 64, DMODEL / 64), 128, 0, stream>>>(
        big, wbuf, f2_b + l * DMODEL, nullptr, h, MTOK, DMODEL, DFFN);
  }

  // --- logits ---
  cvt_kernel<<<2048, 256, 0, stream>>>(h, g16, MTOK * DMODEL);
  cvt_kernel<<<256, 256, 0, stream>>>(out_w, wbuf, VOCAB * DMODEL);
  gemm_kernel<EPI_F32><<<dim3(MTOK / 64, (VOCAB + 63) / 64), 128, 0, stream>>>(
      g16, wbuf, nullptr, nullptr, (float*)d_out, MTOK, VOCAB, DMODEL);
}
